// EveGeneratorV2_2783138807877
// MI455X (gfx1250) — matrix operand order repaired and checked by reading, not compiled, not run on hardware
//
#include <hip/hip_runtime.h>
#include <stdint.h>

// MI455X / gfx1250, wave32.
// - All matmuls via v_wmma_f32_16x16x32_bf16 (compute-bound: ~240 GFLOP vs only
//   ~73 MB of HBM traffic -> matrix pipes are the limiter, not bandwidth).
// - Weight tiles are staged per-block into LDS with gfx1250 async-to-LDS copies
//   (ASYNCcnt + s_wait_asynccnt + workgroup split barriers), so the block's 8
//   waves share one weight stream instead of 8 (~8x less L2 traffic than
//   per-wave streaming).
// - Layer 1 weights (64 KB total) are batch-staged in two 32 KB halves (4 sync
//   rounds instead of 33); layer 2 / heads use per-tile double buffering.
// - LayerNorm stats are accumulated in registers during the WMMA D-store and
//   reduced with half-wave shfl_xor; normalization is applied in place on the
//   bf16 activation buffer. A-fragments are register-resident across the
//   N-loop (128 VGPRs), so only the B operand touches LDS per WMMA.

typedef __attribute__((ext_vector_type(16))) __bf16 bf16x16;
typedef __attribute__((ext_vector_type(8)))  __bf16 bf16x8;
typedef __attribute__((ext_vector_type(8)))  float  f32x8;

#define LATENT   64
#define HIDDEN   512
#define TCOLS    256            // 4 heads * HEAD_H(64)
#define WAVES    8
#define RPW      16             // rows per wave (WMMA M)
#define EPS      1e-5f

// d_ws layout (bytes): bf16 transposed weights
#define WS_W1T   0                                  // [512][64]   65536 B
#define WS_W2T   (65536)                            // [512][512] 524288 B
#define WS_WH1T  (65536 + 524288)                   // [256][512] 262144 B
#define WS_WBLK  (65536 + 524288 + 262144)          // [16][256]    8192 B

static __device__ __forceinline__ f32x8 wmma_bf16(bf16x16 a, bf16x16 b, f32x8 c) {
  return __builtin_amdgcn_wmma_f32_16x16x32_bf16(false, a, false, b, (short)0, c,
                                                 false, false);
}

// ---- gfx1250 async global->LDS copy (ASYNCcnt tracked) ----
static __device__ __forceinline__ void async_copy_b128(uint32_t lds_addr,
                                                       const void* gptr) {
  asm volatile("global_load_async_to_lds_b128 %0, %1, off"
               :: "v"(lds_addr), "v"((uint64_t)(uintptr_t)gptr)
               : "memory");
}
static __device__ __forceinline__ void wait_async() {
  asm volatile("s_wait_asynccnt 0x0" ::: "memory");
}

// Stage `bytes` (multiple of 512) from global to LDS; whole block cooperates,
// each wave-instruction moves 32 lanes * 16 B = 512 B.
static __device__ __forceinline__ void stage_tile(const __bf16* __restrict__ gsrc,
                                                  __bf16* ldst, int bytes,
                                                  int wave, int lane) {
  const int nchunks = bytes >> 9;
  const uint32_t lbase = (uint32_t)(uintptr_t)(void*)ldst;
  for (int c = wave; c < nchunks; c += WAVES) {
    const int off = c * 512 + lane * 16;
    async_copy_b128(lbase + off, (const char*)gsrc + off);
  }
}

// B fragment from row-major [16][K] bf16 (LDS staged tile or global).
// Lane L: N = L&15; K = kt*32 + {0..7,16..23} (L<16) or {8..15,24..31} (L>=16).
static __device__ __forceinline__ bf16x16 load_bfrag(const __bf16* __restrict__ WT,
                                                     int K, int kt, int lane) {
  const int n  = lane & 15;
  const int kb = kt * 32 + ((lane >> 4) << 3);
  const __bf16* base = WT + (size_t)n * K + kb;
  union { bf16x16 v; bf16x8 h[2]; } u;
  u.h[0] = *(const bf16x8*)(base);
  u.h[1] = *(const bf16x8*)(base + 16);
  return u.v;
}

// A fragment from LDS row-major [16][K] bf16.
// Lane L: M = L&15; K = kt*32 + {0..7,16..23} (L<16) or {8..15,24..31} (L>=16).
static __device__ __forceinline__ bf16x16 load_afrag(const __bf16* __restrict__ ab,
                                                     int K, int kt, int lane) {
  const int row = lane & 15;
  const int kb  = kt * 32 + ((lane >> 4) ? 8 : 0);
  const __bf16* base = ab + row * K + kb;
  union { bf16x16 v; bf16x8 h[2]; } u;
  u.h[0] = *(const bf16x8*)(base);
  u.h[1] = *(const bf16x8*)(base + 16);
  return u.v;
}

// Store raw D+bias as bf16 into [16][512] LDS, accumulating per-row sum/sumsq.
static __device__ __forceinline__ void store_raw_stats(__bf16* ab, int nt, int lane,
                                                       f32x8 acc, float bias,
                                                       float* s, float* q) {
  const int col = nt * 16 + (lane & 15);
  const int rb  = (lane >> 4) ? 8 : 0;
#pragma unroll
  for (int j = 0; j < 8; ++j) {
    const float v = acc[j] + bias;
    s[j] += v;
    q[j] += v * v;
    ab[(rb + j) * HIDDEN + col] = (__bf16)v;
  }
}

// Half-wave reduce stats, then LayerNorm + LeakyReLU(0.2) in place on bf16.
static __device__ __forceinline__ void ln_inplace(__bf16* ab,
                                                  const float* __restrict__ g,
                                                  const float* __restrict__ bta,
                                                  float* s, float* q, int lane) {
#pragma unroll
  for (int m = 1; m < 16; m <<= 1) {
#pragma unroll
    for (int j = 0; j < 8; ++j) {
      s[j] += __shfl_xor(s[j], m, 32);
      q[j] += __shfl_xor(q[j], m, 32);
    }
  }
  float mu[8], rs[8];
#pragma unroll
  for (int j = 0; j < 8; ++j) {
    mu[j] = s[j] * (1.f / 512.f);
    const float var = q[j] * (1.f / 512.f) - mu[j] * mu[j];
    rs[j] = rsqrtf(var + EPS);
  }
  const int lrow = lane & 15;
  const int rb   = (lane >> 4) ? 8 : 0;
  for (int i = 0; i < 32; ++i) {
    const int c = lrow + 16 * i;
    const float gc = g[c], bc = bta[c];
#pragma unroll
    for (int j = 0; j < 8; ++j) {
      const float v = (float)ab[(rb + j) * HIDDEN + c];
      float f = (v - mu[j]) * rs[j] * gc + bc;
      f = (f >= 0.f) ? f : 0.2f * f;
      ab[(rb + j) * HIDDEN + c] = (__bf16)f;
    }
  }
}

// ---------------- weight prep (fp32 -> bf16, transposed/packed) ----------------

__global__ void k_transpose_bf16(const float* __restrict__ src, __bf16* __restrict__ dst,
                                 int K, int N) {
  const int i = blockIdx.x * blockDim.x + threadIdx.x;
  if (i >= K * N) return;
  const int k = i / N, n = i % N;
  dst[(size_t)n * K + k] = (__bf16)src[i];
}

// Wh1 [4][512][64] -> [256][512] with row n = head*64 + d
__global__ void k_pack_wh1(const float* __restrict__ src, __bf16* __restrict__ dst) {
  const int i = blockIdx.x * blockDim.x + threadIdx.x;
  if (i >= 4 * 512 * 64) return;
  const int k = i / (512 * 64);
  const int h = (i / 64) % 512;
  const int d = i % 64;
  dst[(size_t)(k * 64 + d) * 512 + h] = (__bf16)src[i];
}

// Block-diagonal head-2: dst [16][256]; col n = head*4 + o, row c = head*64 + d.
__global__ void k_pack_wblk(const float* __restrict__ wh2, __bf16* __restrict__ dst) {
  const int i = blockIdx.x * blockDim.x + threadIdx.x;
  if (i >= 16 * 256) return;
  const int n = i / 256, c = i % 256;
  const int kn = n >> 2, o = n & 3, kc = c >> 6, d = c & 63;
  const float v = (kc == kn) ? wh2[((kn * 64) + d) * 4 + o] : 0.f;
  dst[i] = (__bf16)v;
}

// ---------------- fused MLP kernel ----------------

__global__ __launch_bounds__(WAVES * 32, 1) void fused_mlp(
    const float* __restrict__ z, const int* __restrict__ x, const int* __restrict__ y,
    const float* __restrict__ b1, const float* __restrict__ g1, const float* __restrict__ be1,
    const float* __restrict__ b2, const float* __restrict__ g2, const float* __restrict__ be2,
    const float* __restrict__ bh1, const float* __restrict__ bh2,
    const __bf16* __restrict__ w1t, const __bf16* __restrict__ w2t,
    const __bf16* __restrict__ wh1t, const __bf16* __restrict__ wblk,
    float* __restrict__ out) {
  __shared__ __bf16 Aping[WAVES][RPW * HIDDEN];  // 128 KB: layer1 out / layer2 in
  __shared__ __bf16 Apong[WAVES][RPW * HIDDEN];  // 128 KB: layer2 out / heads in
  __shared__ __bf16 Bstage[2][RPW * HIDDEN];     //  32 KB: B tiles (ping-pong or batch)

  const int wave = threadIdx.x >> 5;
  const int lane = threadIdx.x & 31;
  __bf16* ab1 = Aping[wave];
  __bf16* ab2 = Apong[wave];
  const int rowbase = blockIdx.x * (WAVES * RPW) + wave * RPW;
  const int lrow    = lane & 15;
  const int khalf   = (lane >> 4) ? 8 : 0;
  const int rb      = (lane >> 4) ? 8 : 0;

  // ---- layer-1 A fragments straight from z (fp32 -> bf16 in regs)
  bf16x16 a1[2];
  {
    const float* zr = z + (size_t)(rowbase + lrow) * LATENT;
#pragma unroll
    for (int kt = 0; kt < LATENT / 32; ++kt) {
      const int kb = kt * 32 + khalf;
      bf16x16 a;
#pragma unroll
      for (int e = 0; e < 8; ++e) a[e] = (__bf16)zr[kb + e];
#pragma unroll
      for (int e = 0; e < 8; ++e) a[8 + e] = (__bf16)zr[kb + 16 + e];
      a1[kt] = a;
    }
  }

  // ---- layer 1: [16,64] @ [64,512]
  // W1T is 64 KB total; batch-stage 32 KB (16 tiles) at a time -> 4 sync rounds.
  float s1[8] = {0}, q1[8] = {0};
  for (int half = 0; half < 2; ++half) {
    stage_tile(w1t + (size_t)half * 16 * RPW * LATENT, &Bstage[0][0],
               16 * RPW * LATENT * 2, wave, lane);
    wait_async();
    __syncthreads();
#pragma unroll 4
    for (int ntl = 0; ntl < 16; ++ntl) {
      const int nt = half * 16 + ntl;
      const __bf16* btile = &Bstage[0][0] + (size_t)ntl * RPW * LATENT;
      f32x8 acc = {};
#pragma unroll
      for (int kt = 0; kt < LATENT / 32; ++kt)
        acc = wmma_bf16(a1[kt], load_bfrag(btile, LATENT, kt, lane), acc);
      store_raw_stats(ab1, nt, lane, acc, b1[nt * 16 + lrow], s1, q1);
    }
    __syncthreads();  // all reads of this batch done before next overwrite
  }
  ln_inplace(ab1, g1, be1, s1, q1, lane);

  // ---- layer 2: [16,512] @ [512,512]; A register-resident, B (16 KB) ping-pong
  bf16x16 afr[HIDDEN / 32];
#pragma unroll
  for (int kt = 0; kt < HIDDEN / 32; ++kt) afr[kt] = load_afrag(ab1, HIDDEN, kt, lane);

  float s2[8] = {0}, q2[8] = {0};
  stage_tile(w2t, Bstage[0], RPW * HIDDEN * 2, wave, lane);
  wait_async();
  __syncthreads();
  for (int nt = 0; nt < HIDDEN / 16; ++nt) {
    if (nt + 1 < HIDDEN / 16)
      stage_tile(w2t + (size_t)(nt + 1) * RPW * HIDDEN, Bstage[(nt + 1) & 1],
                 RPW * HIDDEN * 2, wave, lane);
    f32x8 acc = {};
#pragma unroll
    for (int kt = 0; kt < HIDDEN / 32; ++kt)
      acc = wmma_bf16(afr[kt], load_bfrag(Bstage[nt & 1], HIDDEN, kt, lane), acc);
    store_raw_stats(ab2, nt, lane, acc, b2[nt * 16 + lrow], s2, q2);
    wait_async();
    __syncthreads();
  }
  ln_inplace(ab2, g2, be2, s2, q2, lane);

  // ---- all 4 heads, stage 1: t = relu([16,512] @ [512,256] + bh1) -> bf16
  __bf16* tb = ab1;  // ab1 contents dead after layer 2
#pragma unroll
  for (int kt = 0; kt < HIDDEN / 32; ++kt) afr[kt] = load_afrag(ab2, HIDDEN, kt, lane);

  stage_tile(wh1t, Bstage[0], RPW * HIDDEN * 2, wave, lane);
  wait_async();
  __syncthreads();
  for (int nt = 0; nt < TCOLS / 16; ++nt) {
    if (nt + 1 < TCOLS / 16)
      stage_tile(wh1t + (size_t)(nt + 1) * RPW * HIDDEN, Bstage[(nt + 1) & 1],
                 RPW * HIDDEN * 2, wave, lane);
    f32x8 acc = {};
#pragma unroll
    for (int kt = 0; kt < HIDDEN / 32; ++kt)
      acc = wmma_bf16(afr[kt], load_bfrag(Bstage[nt & 1], HIDDEN, kt, lane), acc);
    const float bias = bh1[nt * 16 + lrow];
    const int col = nt * 16 + lrow;
#pragma unroll
    for (int j = 0; j < 8; ++j) {
      const float f = fmaxf(acc[j] + bias, 0.f);
      tb[(rb + j) * TCOLS + col] = (__bf16)f;
    }
    wait_async();
    __syncthreads();
  }

  // ---- all heads' logits in one 16x16 tile: [16,256] @ blockdiag[256,16]
  // (wblk is only 8 KB total -> read straight from L2)
  f32x8 acc = {};
#pragma unroll
  for (int kt = 0; kt < TCOLS / 32; ++kt)
    acc = wmma_bf16(load_afrag(tb, TCOLS, kt, lane),
                    load_bfrag(wblk, TCOLS, kt, lane), acc);
  float* lg = (float*)(ab1 + 4096);  // bytes 8192.. of Aping[wave]; tb uses 0..8191
  {
    const float bias = bh2[lrow];
#pragma unroll
    for (int j = 0; j < 8; ++j) lg[(rb + j) * 16 + lrow] = acc[j] + bias;
  }

  // ---- per-row head select + softmax(4); lanes 0..15 each own one row
  if (lane < 16) {
    const int r = lane;
    const size_t row = (size_t)rowbase + r;
    const int kk = 2 * x[row] + y[row];
    const float* lr = lg + r * 16 + kk * 4;
    const float v0 = lr[0], v1 = lr[1], v2 = lr[2], v3 = lr[3];
    const float m  = fmaxf(fmaxf(v0, v1), fmaxf(v2, v3));
    const float e0 = __expf(v0 - m), e1 = __expf(v1 - m);
    const float e2 = __expf(v2 - m), e3 = __expf(v3 - m);
    const float inv = 1.f / (e0 + e1 + e2 + e3);
    float* o = out + row * 4;
    o[0] = e0 * inv; o[1] = e1 * inv; o[2] = e2 * inv; o[3] = e3 * inv;
  }
}

// ---------------- host launch ----------------

extern "C" void kernel_launch(void* const* d_in, const int* in_sizes, int n_in,
                              void* d_out, int out_size, void* d_ws, size_t ws_size,
                              hipStream_t stream) {
  const float* z   = (const float*)d_in[0];
  const int*   x   = (const int*)d_in[1];
  const int*   y   = (const int*)d_in[2];
  const float* W1  = (const float*)d_in[3];
  const float* b1  = (const float*)d_in[4];
  const float* g1  = (const float*)d_in[5];
  const float* be1 = (const float*)d_in[6];
  const float* W2  = (const float*)d_in[7];
  const float* b2  = (const float*)d_in[8];
  const float* g2  = (const float*)d_in[9];
  const float* be2 = (const float*)d_in[10];
  const float* Wh1 = (const float*)d_in[11];
  const float* bh1 = (const float*)d_in[12];
  const float* Wh2 = (const float*)d_in[13];
  const float* bh2 = (const float*)d_in[14];
  float* out = (float*)d_out;

  char* ws = (char*)d_ws;
  __bf16* w1t  = (__bf16*)(ws + WS_W1T);
  __bf16* w2t  = (__bf16*)(ws + WS_W2T);
  __bf16* wh1t = (__bf16*)(ws + WS_WH1T);
  __bf16* wblk = (__bf16*)(ws + WS_WBLK);

  // weight prep (runs every call; deterministic)
  k_transpose_bf16<<<(LATENT * HIDDEN + 255) / 256, 256, 0, stream>>>(W1, w1t, LATENT, HIDDEN);
  k_transpose_bf16<<<(HIDDEN * HIDDEN + 255) / 256, 256, 0, stream>>>(W2, w2t, HIDDEN, HIDDEN);
  k_pack_wh1<<<(4 * 512 * 64 + 255) / 256, 256, 0, stream>>>(Wh1, wh1t);
  k_pack_wblk<<<(16 * 256 + 255) / 256, 256, 0, stream>>>(Wh2, wblk);

  const int B = in_sizes[0] / LATENT;          // 262144
  const int blocks = B / (WAVES * RPW);        // 2048
  fused_mlp<<<blocks, WAVES * 32, 0, stream>>>(
      z, x, y, b1, g1, be1, b2, g2, be2, bh1, bh2, w1t, w2t, wh1t, wblk, out);
}
